// PolicyNetwork_53326313947485
// MI455X (gfx1250) — matrix operand order repaired and checked by reading, not compiled, not run on hardware
//
#include <hip/hip_runtime.h>
#include <stdint.h>

#define DIM 128

typedef __attribute__((ext_vector_type(16))) __bf16 v16bf;
typedef __attribute__((ext_vector_type(8)))  float  v8f;

union BfVec { v16bf v; uint4 q[2]; };

__device__ __forceinline__ uint16_t f2bf(float f) {
    uint32_t u = __float_as_uint(f);
    uint32_t r = u + 0x7FFFu + ((u >> 16) & 1u);   // round-to-nearest-even
    return (uint16_t)(r >> 16);
}

// ---------------- utility kernels ----------------
__global__ void k_zero(float* __restrict__ p, int n) {
    int i = blockIdx.x * blockDim.x + threadIdx.x;
    if (i < n) p[i] = 0.f;
}

__global__ void k_deg(const int* __restrict__ dst, float* __restrict__ deg, int nE) {
    int e = blockIdx.x * blockDim.x + threadIdx.x;
    if (e < nE) atomicAdd(&deg[dst[e]], 1.0f);
}

__global__ void k_dinv(float* __restrict__ deg, int n) {
    int i = blockIdx.x * blockDim.x + threadIdx.x;
    if (i < n) deg[i] = rsqrtf(1.0f + deg[i]);     // in place: deg -> dinv
}

__global__ void k_cvt(const float* __restrict__ x, uint16_t* __restrict__ xb, int n) {
    int i = blockIdx.x * blockDim.x + threadIdx.x;
    if (i < n) xb[i] = f2bf(x[i]);
}

// ---------------- bf16 WMMA GEMM: h[N x 128] = xb[N x 128] @ W[128 x 128] ----------------
__global__ __launch_bounds__(256) void k_gemm(const uint16_t* __restrict__ xb,
                                              const float* __restrict__ W,
                                              float* __restrict__ h, int nrows) {
    __shared__ __align__(16) uint16_t Wt[DIM * 136];   // transposed W, bf16, padded pitch
    const int tid = threadIdx.x;
    for (int i = tid; i < DIM * DIM; i += 256) {
        int k = i >> 7, n = i & 127;                   // W row-major [k][n]
        Wt[n * 136 + k] = f2bf(W[i]);                  // store as Wt[n][k]
    }
    __syncthreads();

    const int wave = tid >> 5, lane = tid & 31;
    const int row0 = (blockIdx.x * 8 + wave) * 16;
    if (row0 >= nrows) return;                         // wave-uniform: EXEC all-ones inside

    const int hi   = lane >> 4;                        // half-wave select
    const int lrow = row0 + (lane & 15);

    v8f acc[8];
#pragma unroll
    for (int nt = 0; nt < 8; ++nt) {
        v8f z = {0.f, 0.f, 0.f, 0.f, 0.f, 0.f, 0.f, 0.f};
        acc[nt] = z;
    }

#pragma unroll
    for (int kt = 0; kt < 4; ++kt) {
        // A tile 16x32 bf16: lanes 0-15 hold K {0..7,16..23}, lanes 16-31 hold K {8..15,24..31}
        BfVec a;
        const uint16_t* ap = xb + lrow * DIM + kt * 32 + hi * 8;
        a.q[0] = *(const uint4*)(ap);
        a.q[1] = *(const uint4*)(ap + 16);
#pragma unroll
        for (int nt = 0; nt < 8; ++nt) {
            // B tile 32x16 bf16: col = nt*16 + lane%16; lanes 0-15: K {0..7,16..23}, lanes 16-31: K {8..15,24..31}
            BfVec b;
            const uint16_t* bp = &Wt[(nt * 16 + (lane & 15)) * 136 + kt * 32 + hi * 8];
            b.q[0] = *(const uint4*)(bp);
            b.q[1] = *(const uint4*)(bp + 16);
            acc[nt] = __builtin_amdgcn_wmma_f32_16x16x32_bf16(
                false, a.v, false, b.v, (short)0, acc[nt], false, false);
        }
    }

#pragma unroll
    for (int nt = 0; nt < 8; ++nt) {
#pragma unroll
        for (int v = 0; v < 8; ++v) {
            int m = v + hi * 8;                         // C/D layout: VGPR v -> row M
            h[(row0 + m) * DIM + nt * 16 + (lane & 15)] = acc[nt][v];
        }
    }
}

// ---------------- agg = h * dinv^2 + bias (self-loop term) ----------------
__global__ void k_initagg(const float* __restrict__ h, const float* __restrict__ dinv,
                          const float* __restrict__ b, float* __restrict__ agg, int nN) {
    int i = blockIdx.x * blockDim.x + threadIdx.x;      // over nN*32 float4 groups
    if (i >= nN * 32) return;
    int node = i >> 5;
    int c4 = (i & 31) * 4;
    float d = dinv[node];
    float dd = d * d;
    const float4 hv = *(const float4*)(h + node * DIM + c4);
    const float4 bv = *(const float4*)(b + c4);
    float4 o;
    o.x = hv.x * dd + bv.x; o.y = hv.y * dd + bv.y;
    o.z = hv.z * dd + bv.z; o.w = hv.w * dd + bv.w;
    *(float4*)(agg + node * DIM + c4) = o;
}

// ---------------- edge scatter-add: agg[dst] += h[src] * dinv[src]*dinv[dst] ----------------
__global__ void k_scatter(const int* __restrict__ src, const int* __restrict__ dst,
                          const float* __restrict__ h, const float* __restrict__ dinv,
                          float* __restrict__ agg, int nE) {
    int t = blockIdx.x * blockDim.x + threadIdx.x;
    int e = t >> 5;                                     // one wave per edge
    if (e >= nE) return;
    int lane = t & 31;
    int s = src[e], d = dst[e];
    float w = dinv[s] * dinv[d];
    const float4 hv = *(const float4*)(h + s * DIM + lane * 4);
    float* ap = agg + d * DIM + lane * 4;
    atomicAdd(ap + 0, hv.x * w);
    atomicAdd(ap + 1, hv.y * w);
    atomicAdd(ap + 2, hv.z * w);
    atomicAdd(ap + 3, hv.w * w);
}

// ---------------- relu in place + bf16 copy for next layer's GEMM ----------------
__global__ void k_relu_cvt(float* __restrict__ agg, uint16_t* __restrict__ xb, int n) {
    int i = blockIdx.x * blockDim.x + threadIdx.x;
    if (i >= n) return;
    float v = fmaxf(agg[i], 0.f);
    agg[i] = v;
    xb[i] = f2bf(v);
}

// ---------------- mean pool partial sums ----------------
__global__ void k_poolsum(const float* __restrict__ agg, float* __restrict__ pooled, int nN) {
    int t = blockIdx.x * blockDim.x + threadIdx.x;      // 512*256 = 131072 threads
    int c = t & 127;
    int r0 = t >> 7;                                    // 1024 row-slices
    float acc = 0.f;
    for (int r = r0; r < nN; r += 1024) acc += agg[r * DIM + c];
    atomicAdd(&pooled[c], acc);
}

// ---------------- head: mean, FC 128->4, softmax ----------------
__global__ void k_head(const float* __restrict__ pooled, const float* __restrict__ fcw,
                       const float* __restrict__ fcb, float* __restrict__ out, int nN) {
    __shared__ float lg[4];
    int c = threadIdx.x;                                // 128 threads
    if (c < 4) lg[c] = 0.f;
    __syncthreads();
    float p = pooled[c] / (float)nN;
#pragma unroll
    for (int j = 0; j < 4; ++j) atomicAdd(&lg[j], p * fcw[c * 4 + j]);
    __syncthreads();
    if (c == 0) {
        float l[4], m = -3.402823466e+38f;
#pragma unroll
        for (int j = 0; j < 4; ++j) { l[j] = lg[j] + fcb[j]; m = fmaxf(m, l[j]); }
        float s = 0.f;
#pragma unroll
        for (int j = 0; j < 4; ++j) { l[j] = __expf(l[j] - m); s += l[j]; }
        float inv = 1.0f / s;
#pragma unroll
        for (int j = 0; j < 4; ++j) out[j] = l[j] * inv;
    }
}

extern "C" void kernel_launch(void* const* d_in, const int* in_sizes, int n_in,
                              void* d_out, int out_size, void* d_ws, size_t ws_size,
                              hipStream_t stream) {
    (void)n_in; (void)out_size; (void)ws_size;

    const float* x   = (const float*)d_in[0];
    const int*   ei  = (const int*)d_in[1];
    const float* Wl[3] = {(const float*)d_in[2], (const float*)d_in[4], (const float*)d_in[6]};
    const float* bl[3] = {(const float*)d_in[3], (const float*)d_in[5], (const float*)d_in[7]};
    const float* fcw = (const float*)d_in[8];
    const float* fcb = (const float*)d_in[9];

    const int nN = in_sizes[0] / DIM;       // 100000
    const int nE = in_sizes[1] / 2;         // 1600000
    const int* src = ei;
    const int* dst = ei + nE;

    // workspace layout (bytes, 256-aligned)
    char* ws = (char*)d_ws;
    const size_t feat_f32 = (size_t)nN * DIM * sizeof(float);      // 51.2 MB
    const size_t feat_b16 = (size_t)nN * DIM * sizeof(uint16_t);   // 25.6 MB
    float*    h      = (float*)(ws);
    float*    agg    = (float*)(ws + feat_f32);
    uint16_t* xb     = (uint16_t*)(ws + 2 * feat_f32);
    float*    dinv   = (float*)(ws + 2 * feat_f32 + feat_b16);
    float*    pooled = (float*)(ws + 2 * feat_f32 + feat_b16 + ((size_t)nN * 4 + 255 & ~(size_t)255));

    const int nF = nN * DIM;                 // 12.8M feature elements

    // degree -> dinv (deterministic re-init every call)
    k_zero<<<(nN + 255) / 256, 256, 0, stream>>>(dinv, nN);
    k_deg<<<(nE + 255) / 256, 256, 0, stream>>>(dst, dinv, nE);
    k_dinv<<<(nN + 255) / 256, 256, 0, stream>>>(dinv, nN);

    // bf16 input features
    k_cvt<<<(nF + 255) / 256, 256, 0, stream>>>(x, xb, nF);

    const int gemmBlocks = (nN / 16 + 7) / 8;
    for (int l = 0; l < 3; ++l) {
        k_gemm<<<gemmBlocks, 256, 0, stream>>>(xb, Wl[l], h, nN);
        k_initagg<<<(nN * 32 + 255) / 256, 256, 0, stream>>>(h, dinv, bl[l], agg, nN);
        k_scatter<<<((size_t)nE * 32 + 255) / 256, 256, 0, stream>>>(src, dst, h, dinv, agg, nE);
        k_relu_cvt<<<(nF + 255) / 256, 256, 0, stream>>>(agg, xb, nF);
    }

    k_zero<<<1, 128, 0, stream>>>(pooled, DIM);
    k_poolsum<<<512, 256, 0, stream>>>(agg, pooled, nN);
    k_head<<<1, 128, 0, stream>>>(pooled, fcw, fcb, (float*)d_out, nN);
}
